// NCL_50766513438746
// MI455X (gfx1250) — matrix operand order repaired and checked by reading, not compiled, not run on hardware
//
#include <hip/hip_runtime.h>

// MI455X / gfx1250 fused MLP forward + input-gradient kernel, v3.
// f16 WMMA (v_wmma_f32_16x16x32_f16) for all 512-wide GEMMs.
// v3: same blocking as v2 (32 samples/WG, B-fragments reused across two
// 16-row M-groups -> halved L2 weight traffic), but B is hoisted in two
// half-K batches of 8 fragments (64 VGPRs) with both group accumulators
// kept live, so the wave stays under 256 VGPRs (no s_set_vgpr_msb churn,
// guaranteed launchable at 8 waves/WG). 128KB LDS (4 slots of 32x512 f16).

typedef __attribute__((ext_vector_type(16))) _Float16 v16h;
typedef __attribute__((ext_vector_type(8)))  float    v8f;

#define HID     512
#define MW      32          // samples per workgroup (2 groups of 16)
#define GROUPS  2
#define THREADS 256         // 8 waves of 32
#define NTPW    4           // N-tiles per wave (32 tiles / 8 waves)
#define NBLK    (131072 / MW)

union V16 { uint4 u[2]; v16h h; };

__device__ __forceinline__ float sp20(float z) {
    // softplus(20*z)/20, numerically stable
    float t = 20.f * z;
    if (t > 15.f) return z;
    return __logf(1.f + __expf(t)) * 0.05f;
}
__device__ __forceinline__ float sig20(float h) {
    // sigmoid(20*z) recovered from h = softplus(20z)/20 :  1 - exp(-20h)
    return 1.f - __expf(-20.f * h);
}

__device__ __forceinline__ V16 loadA(const _Float16* p) {
    V16 A;
    A.u[0] = *(const uint4*)(p);
    A.u[1] = *(const uint4*)(p + 8);
    return A;
}

// ---- forward hidden layer:  out = softplus20( in @ W^T + b ) -------------
__device__ __forceinline__ void fwd_layer(const _Float16* __restrict__ in,
                                          _Float16* __restrict__ outp,
                                          const _Float16* __restrict__ W,   // [512][512] row-major (dout,din)
                                          const float* __restrict__ bias,
                                          int lane, int wave)
{
    const int m     = lane & 15;
    const int koffA = (lane >> 4) << 4;   // A layout: one 16-half run
    const int koffB = (lane >> 4) << 4;   // B layout: one 16-half run
    const int mbase = (lane >> 4) << 3;

    for (int t = 0; t < NTPW; ++t) {
        const int n = (wave * NTPW + t) * 16 + (lane & 15);
        const _Float16* wr  = W + (size_t)n * HID + koffB;
        const _Float16* a0p = in + m * HID + koffA;
        const _Float16* a1p = a0p + 16 * HID;
        const float bn = bias[n];
        v8f c0 = {}, c1 = {};
        #pragma unroll
        for (int kh = 0; kh < 2; ++kh) {            // two half-K batches
            V16 Bf[8];                              // hoisted: 64 VGPRs
            #pragma unroll
            for (int kk = 0; kk < 8; ++kk) {
                const int kc = kh * 8 + kk;
                Bf[kk].u[0] = *(const uint4*)(wr + kc * 32);
                Bf[kk].u[1] = *(const uint4*)(wr + kc * 32 + 8);
            }
            #pragma unroll
            for (int kk = 0; kk < 8; ++kk) {        // group 0 reuses Bf
                const int kc = kh * 8 + kk;
                V16 A = loadA(a0p + kc * 32);
                c0 = __builtin_amdgcn_wmma_f32_16x16x32_f16(false, A.h, false, Bf[kk].h,
                                                            (short)0, c0, false, false);
            }
            #pragma unroll
            for (int kk = 0; kk < 8; ++kk) {        // group 1 reuses Bf
                const int kc = kh * 8 + kk;
                V16 A = loadA(a1p + kc * 32);
                c1 = __builtin_amdgcn_wmma_f32_16x16x32_f16(false, A.h, false, Bf[kk].h,
                                                            (short)0, c1, false, false);
            }
        }
        #pragma unroll
        for (int r = 0; r < 8; ++r) {
            outp[(mbase + r) * HID + n]            = (_Float16)sp20(c0[r] + bn);
            outp[16 * HID + (mbase + r) * HID + n] = (_Float16)sp20(c1[r] + bn);
        }
    }
}

// ---- backward layer:  sig_io = (inA @ W) * sigma(sig_io)  (in place) -----
__device__ __forceinline__ void bwd_layer(const _Float16* __restrict__ inA,
                                          _Float16* __restrict__ sig_io,
                                          const _Float16* __restrict__ WT,  // [din][dout] = W transposed
                                          int lane, int wave)
{
    const int m     = lane & 15;
    const int koffA = (lane >> 4) << 4;
    const int koffB = (lane >> 4) << 4;
    const int mbase = (lane >> 4) << 3;

    for (int t = 0; t < NTPW; ++t) {
        const int n = (wave * NTPW + t) * 16 + (lane & 15);
        const _Float16* wr  = WT + (size_t)n * HID + koffB;
        const _Float16* a0p = inA + m * HID + koffA;
        const _Float16* a1p = a0p + 16 * HID;
        v8f c0 = {}, c1 = {};
        #pragma unroll
        for (int kh = 0; kh < 2; ++kh) {
            V16 Bf[8];
            #pragma unroll
            for (int kk = 0; kk < 8; ++kk) {
                const int kc = kh * 8 + kk;
                Bf[kk].u[0] = *(const uint4*)(wr + kc * 32);
                Bf[kk].u[1] = *(const uint4*)(wr + kc * 32 + 8);
            }
            #pragma unroll
            for (int kk = 0; kk < 8; ++kk) {
                const int kc = kh * 8 + kk;
                V16 A = loadA(a0p + kc * 32);
                c0 = __builtin_amdgcn_wmma_f32_16x16x32_f16(false, A.h, false, Bf[kk].h,
                                                            (short)0, c0, false, false);
            }
            #pragma unroll
            for (int kk = 0; kk < 8; ++kk) {
                const int kc = kh * 8 + kk;
                V16 A = loadA(a1p + kc * 32);
                c1 = __builtin_amdgcn_wmma_f32_16x16x32_f16(false, A.h, false, Bf[kk].h,
                                                            (short)0, c1, false, false);
            }
        }
        #pragma unroll
        for (int r = 0; r < 8; ++r) {
            const int i0 = (mbase + r) * HID + n;
            const int i1 = 16 * HID + i0;
            const float sA = sig20((float)sig_io[i0]);
            sig_io[i0] = (_Float16)(c0[r] * sA);
            const float sB = sig20((float)sig_io[i1]);
            sig_io[i1] = (_Float16)(c1[r] * sB);
        }
    }
}

// ---- weight repack kernels ----------------------------------------------
__global__ void prep_big(const float* __restrict__ W1, const float* __restrict__ W2,
                         const float* __restrict__ W3, _Float16* __restrict__ ws)
{
    int idx = blockIdx.x * blockDim.x + threadIdx.x;
    if (idx >= 3 * HID * HID) return;
    const int l = idx / (HID * HID);
    const int e = idx - l * (HID * HID);
    const int n = e >> 9, k = e & 511;
    const float* Wsrc = (l == 0) ? W1 : (l == 1) ? W2 : W3;
    const float v = Wsrc[e];
    ws[(size_t)l * HID * HID + e] = (_Float16)v;                         // row-major (fwd)
    ws[(size_t)(3 + l) * HID * HID + (size_t)k * HID + n] = (_Float16)v; // transposed (bwd)
}

__global__ void prep_small(const float* __restrict__ W0, const float* __restrict__ W4,
                           _Float16* __restrict__ ws)
{
    int idx = blockIdx.x * blockDim.x + threadIdx.x;
    const size_t W0T_off = (size_t)6 * HID * HID;
    if (idx < 16 * HID) {                       // W0^T zero-padded to 16x512
        const int r = idx >> 9, k = idx & 511;
        ws[W0T_off + idx] = (r < 3) ? (_Float16)W0[k * 3 + r] : (_Float16)0.f;
    } else if (idx < 16 * HID + HID) {          // W4 row 1 (pressure weights)
        const int k = idx - 16 * HID;
        ws[W0T_off + 16 * HID + k] = (_Float16)W4[HID + k];
    }
}

// ---- main fused kernel ---------------------------------------------------
__global__ __launch_bounds__(THREADS)
void mlp_fused(const float* __restrict__ tx,
               const float* __restrict__ W0, const float* __restrict__ b0,
               const float* __restrict__ b1, const float* __restrict__ b2,
               const float* __restrict__ b3,
               const float* __restrict__ W4, const float* __restrict__ b4,
               const _Float16* __restrict__ Wf1, const _Float16* __restrict__ Wf2,
               const _Float16* __restrict__ Wf3,
               const _Float16* __restrict__ WT1, const _Float16* __restrict__ WT2,
               const _Float16* __restrict__ WT3,
               const _Float16* __restrict__ W0T, const _Float16* __restrict__ W4r1,
               float* __restrict__ out)
{
    __shared__ __align__(16) _Float16 lds[4 * MW * HID];   // 128 KB: h0..h3 / d-slots
    _Float16* s0 = lds;
    _Float16* s1 = lds + 1 * MW * HID;
    _Float16* s2 = lds + 2 * MW * HID;
    _Float16* s3 = lds + 3 * MW * HID;

    const int tid  = threadIdx.x;
    const int lane = tid & 31;
    const int wave = tid >> 5;
    const int base = blockIdx.x * MW;

    // ---- layer 0 (K=3): VALU, trivial ----
    for (int e = tid; e < MW * HID; e += THREADS) {
        const int m = e >> 9, n = e & 511;
        const float* x = tx + (size_t)(base + m) * 3;
        const float z = b0[n] + x[0] * W0[n * 3 + 0] + x[1] * W0[n * 3 + 1]
                              + x[2] * W0[n * 3 + 2];
        s0[e] = (_Float16)sp20(z);
    }
    __syncthreads();

    // ---- forward hidden layers (WMMA) ----
    fwd_layer(s0, s1, Wf1, b1, lane, wave);  __syncthreads();
    fwd_layer(s1, s2, Wf2, b2, lane, wave);  __syncthreads();
    fwd_layer(s2, s3, Wf3, b3, lane, wave);  __syncthreads();

    // ---- pressure: p = h3 . W4[1] + b4[1]  (waves 0-1, one per M-group) ----
    if (wave < GROUPS) {
        const int g     = wave;
        const int m     = lane & 15;
        const int koffA = (lane >> 4) << 4;
        const int koffB = (lane >> 4) << 4;
        const int mbase = (lane >> 4) << 3;
        const _Float16* s3g = s3 + g * 16 * HID;
        v8f C = {};
        #pragma unroll
        for (int kc = 0; kc < 16; ++kc) {
            V16 A = loadA(s3g + m * HID + kc * 32 + koffA);
            V16 Bp;
            uint4 z4; z4.x = z4.y = z4.z = z4.w = 0u;
            Bp.u[0] = z4; Bp.u[1] = z4;
            if ((lane & 15) == 0) {
                const _Float16* q = W4r1 + kc * 32 + koffB;
                Bp.u[0] = *(const uint4*)(q);
                Bp.u[1] = *(const uint4*)(q + 8);
            }
            C = __builtin_amdgcn_wmma_f32_16x16x32_f16(false, A.h, false, Bp.h,
                                                       (short)0, C, false, false);
        }
        if ((lane & 15) == 0) {
            const float b41 = b4[1];
            #pragma unroll
            for (int r = 0; r < 8; ++r)
                out[(size_t)(base + g * 16 + mbase + r) * 3 + 2] = C[r] + b41;
        }
    }
    __syncthreads();

    // ---- d3 = W4[0] * sigma(h3), in place in s3 ----
    for (int e = tid; e < MW * HID; e += THREADS) {
        const int n = e & 511;
        const float h = (float)s3[e];
        s3[e] = (_Float16)(W4[n] * sig20(h));
    }
    __syncthreads();

    // ---- backward chain (WMMA) ----
    bwd_layer(s3, s2, WT3, lane, wave);  __syncthreads();   // d2
    bwd_layer(s2, s1, WT2, lane, wave);  __syncthreads();   // d1
    bwd_layer(s1, s0, WT1, lane, wave);  __syncthreads();   // d0

    // ---- g = d0 @ W0  (N=3 padded to 16; waves 0-1, one per M-group) ----
    if (wave < GROUPS) {
        const int g     = wave;
        const int m     = lane & 15;
        const int koffA = (lane >> 4) << 4;
        const int koffB = (lane >> 4) << 4;
        const int mbase = (lane >> 4) << 3;
        const int n0    = lane & 15;
        const _Float16* s0g = s0 + g * 16 * HID;
        v8f C = {};
        #pragma unroll
        for (int kc = 0; kc < 16; ++kc) {
            V16 A = loadA(s0g + m * HID + kc * 32 + koffA);
            V16 B;
            const _Float16* q = W0T + (size_t)n0 * HID + kc * 32 + koffB;  // rows >=3 zero
            B.u[0] = *(const uint4*)(q);
            B.u[1] = *(const uint4*)(q + 8);
            C = __builtin_amdgcn_wmma_f32_16x16x32_f16(false, A.h, false, B.h,
                                                       (short)0, C, false, false);
        }
        #pragma unroll
        for (int r = 0; r < 8; ++r) {
            const int mm = base + g * 16 + mbase + r;
            if (n0 == 2) out[(size_t)mm * 3 + 0] =  C[r];   // u_x =  ds/dy
            if (n0 == 1) out[(size_t)mm * 3 + 1] = -C[r];   // u_y = -ds/dx
        }
    }
}

extern "C" void kernel_launch(void* const* d_in, const int* in_sizes, int n_in,
                              void* d_out, int out_size, void* d_ws, size_t ws_size,
                              hipStream_t stream) {
    const float* tx = (const float*)d_in[0];
    const float* W0 = (const float*)d_in[1];
    const float* b0 = (const float*)d_in[2];
    const float* W1 = (const float*)d_in[3];
    const float* b1 = (const float*)d_in[4];
    const float* W2 = (const float*)d_in[5];
    const float* b2 = (const float*)d_in[6];
    const float* W3 = (const float*)d_in[7];
    const float* b3 = (const float*)d_in[8];
    const float* W4 = (const float*)d_in[9];
    const float* b4 = (const float*)d_in[10];
    float* out = (float*)d_out;

    _Float16* ws  = (_Float16*)d_ws;          // needs ~3.2 MB
    _Float16* Wf1 = ws + (size_t)0 * HID * HID;
    _Float16* Wf2 = ws + (size_t)1 * HID * HID;
    _Float16* Wf3 = ws + (size_t)2 * HID * HID;
    _Float16* WT1 = ws + (size_t)3 * HID * HID;
    _Float16* WT2 = ws + (size_t)4 * HID * HID;
    _Float16* WT3 = ws + (size_t)5 * HID * HID;
    _Float16* W0T = ws + (size_t)6 * HID * HID;
    _Float16* W4r1 = W0T + 16 * HID;

    prep_big<<<(3 * HID * HID + 255) / 256, 256, 0, stream>>>(W1, W2, W3, ws);
    prep_small<<<(17 * HID + 255) / 256, 256, 0, stream>>>(W0, W4, ws);

    mlp_fused<<<NBLK, THREADS, 0, stream>>>(tx, W0, b0, b1, b2, b3, W4, b4,
                                            Wf1, Wf2, Wf3, WT1, WT2, WT3,
                                            W0T, W4r1, out);
}
